// Llama4MoE_764504179345
// MI455X (gfx1250) — matrix operand order repaired and checked by reading, not compiled, not run on hardware
//
#include <hip/hip_runtime.h>
#include <hip/hip_bf16.h>

// ---------------- problem constants ----------------
constexpr int TT = 4096;   // tokens
constexpr int HH = 2048;   // hidden
constexpr int II = 1024;   // intermediate
constexpr int EE = 16;     // experts
constexpr int TM = 64;     // M tile (rows)
constexpr int TN = 64;     // N tile (cols)
constexpr int TK = 32;     // K step (bf16 wmma depth)
constexpr int LDA = 40;    // padded LDS row stride (elements), bank-safe
constexpr int LDB = 40;
constexpr int ASZ = TM * LDA;          // one A buffer (elements)
constexpr int BSZ = TN * LDB;          // one B buffer (elements)
constexpr int TPAD = TT + EE * TM;     // 5120 max padded rows
constexpr int NMT  = TPAD / TM;        // 80 routed M-tiles

typedef __attribute__((ext_vector_type(16))) __bf16 v16bf;
typedef __attribute__((ext_vector_type(8)))  __bf16 v8bf;
typedef __attribute__((ext_vector_type(8)))  float  v8f;
typedef __attribute__((ext_vector_type(4)))  float  v4f;
typedef __attribute__((ext_vector_type(4)))  unsigned int v4u;
typedef __attribute__((ext_vector_type(8)))  int v8i;
typedef __attribute__((ext_vector_type(4)))  int v4i;

__device__ __forceinline__ float fast_sigmoid(float x) {
  return __builtin_amdgcn_rcpf(1.0f + __expf(-x));
}

// ---- Tensor Data Mover: load a TM x TK bf16 tile (row-major, row length
// `rowlen` elements) from global into LDS with hardware padding to LDA
// elements per row (16 DWORDs data + 4 DWORDs pad = 80B stride). ----
__device__ __forceinline__ void tdm_load_tile(unsigned lds_addr, const void* gaddr,
                                              unsigned rowlen, unsigned nrows) {
  unsigned long long ga = (unsigned long long)gaddr;
  v4u g0;
  g0[0] = 1u;                                   // count=1 (valid user D#)
  g0[1] = lds_addr;                             // LDS byte address
  g0[2] = (unsigned)ga;                         // global addr lo
  g0[3] = (unsigned)(ga >> 32) | (2u << 30);    // global addr hi | type=2
  v8i g1;
  // data_size=1 (2B), pad_enable=1, pad_interval=3 (16 DW = 64B),
  // pad_amount=3 (4 DW = 16B)
  g1[0] = (int)((1u << 16) | (1u << 20) | (3u << 22) | (3u << 25));
  g1[1] = (int)((rowlen & 0xFFFFu) << 16);                    // tensor_dim0 lo16
  g1[2] = (int)((rowlen >> 16) | ((nrows & 0xFFFFu) << 16));  // dim0 hi | dim1 lo
  g1[3] = (int)((nrows >> 16) | ((unsigned)TK << 16));        // dim1 hi | tile_dim0
  g1[4] = TM;                                                  // tile_dim1 (dim2=0)
  g1[5] = (int)rowlen;                                         // dim0_stride lo
  g1[6] = 0;                                                   // stride hi / dim1_stride
  g1[7] = 0;
  v4i z4 = {0, 0, 0, 0};
  v8i z8 = {0, 0, 0, 0, 0, 0, 0, 0};
  __builtin_amdgcn_tensor_load_to_lds(g0, g1, z4, z4, z8, 0);
}

// ---------------- init scratch state ----------------
__global__ __launch_bounds__(256) void init_kernel(int* counts, int* cursor,
                                                   int* tileE, int* perm) {
  int tid = threadIdx.x;
  if (tid < EE) { counts[tid] = 0; cursor[tid] = 0; }
  for (int i = tid; i < NMT; i += 256) tileE[i] = -1;
  for (int i = tid; i < TPAD; i += 256) perm[i] = -1;
}

// ---------------- router: wave-per-token top-1 + sigmoid ----------------
__global__ __launch_bounds__(256) void router_kernel(const float* __restrict__ X,
                                                     const float* __restrict__ RW,
                                                     int* __restrict__ eid,
                                                     float* __restrict__ wgt) {
  int t    = blockIdx.x * 8 + (threadIdx.x >> 5);
  int lane = threadIdx.x & 31;
  float acc[EE];
#pragma unroll
  for (int e = 0; e < EE; ++e) acc[e] = 0.f;

  for (int k0 = lane * 4; k0 < HH; k0 += 128) {
    v4f xv = *(const v4f*)(X + (long)t * HH + k0);
#pragma unroll
    for (int j = 0; j < 4; ++j) {
      const float* rw = RW + (long)(k0 + j) * EE;
      v4f r0 = *(const v4f*)(rw + 0);
      v4f r1 = *(const v4f*)(rw + 4);
      v4f r2 = *(const v4f*)(rw + 8);
      v4f r3 = *(const v4f*)(rw + 12);
      float x = xv[j];
#pragma unroll
      for (int i = 0; i < 4; ++i) {
        acc[i]      += x * r0[i];
        acc[4 + i]  += x * r1[i];
        acc[8 + i]  += x * r2[i];
        acc[12 + i] += x * r3[i];
      }
    }
  }
#pragma unroll
  for (int e = 0; e < EE; ++e)
    for (int off = 16; off > 0; off >>= 1)
      acc[e] += __shfl_xor(acc[e], off, 32);

  if (lane == 0) {
    int best = 0; float bv = acc[0];
#pragma unroll
    for (int e = 1; e < EE; ++e)
      if (acc[e] > bv) { bv = acc[e]; best = e; }
    eid[t] = best;
    wgt[t] = fast_sigmoid(bv);
  }
}

// ---------------- expert histogram ----------------
__global__ __launch_bounds__(256) void count_kernel(const int* __restrict__ eid,
                                                    int* __restrict__ counts) {
  int t = blockIdx.x * 256 + threadIdx.x;
  atomicAdd(&counts[eid[t]], 1);
}

// ---------------- tile-aligned prefix + tile->expert map ----------------
__global__ void scan_kernel(const int* __restrict__ counts,
                            int* __restrict__ offsets,
                            int* __restrict__ tileE) {
  int off = 0;
  for (int e = 0; e < EE; ++e) {
    offsets[e] = off;
    int padded = (counts[e] + TM - 1) / TM * TM;
    for (int i = 0; i < padded / TM; ++i) tileE[off / TM + i] = e;
    off += padded;
  }
}

// ---------------- scatter token -> sorted slot ----------------
__global__ __launch_bounds__(256) void scatter_kernel(const int* __restrict__ eid,
                                                      const int* __restrict__ offsets,
                                                      int* __restrict__ cursor,
                                                      int* __restrict__ perm) {
  int t = blockIdx.x * 256 + threadIdx.x;
  int e = eid[t];
  int pos = offsets[e] + atomicAdd(&cursor[e], 1);
  perm[pos] = t;
}

// ---------------- bf16 copy of X for shared path ----------------
__global__ __launch_bounds__(256) void cvt_shared_kernel(const float* __restrict__ X,
                                                         __bf16* __restrict__ Xsh) {
  long row = blockIdx.x;
  int i = threadIdx.x * 8;
  v4f a = *(const v4f*)(X + row * HH + i);
  v4f b = *(const v4f*)(X + row * HH + i + 4);
  v8bf o;
#pragma unroll
  for (int j = 0; j < 4; ++j) { o[j] = (__bf16)a[j]; o[4 + j] = (__bf16)b[j]; }
  *(v8bf*)(Xsh + row * HH + i) = o;
}

// ---------------- gather weighted rows (routed path, zero pad rows) ----------
__global__ __launch_bounds__(256) void gather_kernel(const float* __restrict__ X,
                                                     const int* __restrict__ perm,
                                                     const float* __restrict__ wgt,
                                                     __bf16* __restrict__ Xg) {
  long row = blockIdx.x;
  int t = perm[row];
  int i = threadIdx.x * 8;
  __bf16* dst = Xg + row * HH + i;
  if (t < 0) {
    v8bf z = {};
    *(v8bf*)dst = z;
    return;
  }
  float w = wgt[t];
  const float* src = X + (long)t * HH + i;
  v4f a = *(const v4f*)(src);
  v4f b = *(const v4f*)(src + 4);
  v8bf o;
#pragma unroll
  for (int j = 0; j < 4; ++j) {
    o[j]     = (__bf16)(a[j] * w);
    o[4 + j] = (__bf16)(b[j] * w);
  }
  *(v8bf*)dst = o;
}

// ---------------- GEMM1: Y = silu(X@W1) * (X@W3), bf16 WMMA ----------------
// Double-buffered: TDM (TENSORcnt) streams the A tile, B weights go
// global->regs->cvt->LDS, 4 independent WMMAs consume the current buffer.
__global__ __launch_bounds__(256)
void gemm1_kernel(const __bf16* __restrict__ X, const float* __restrict__ W1base,
                  const float* __restrict__ W3base, __bf16* __restrict__ Y,
                  const int* __restrict__ tileE, long wstride, int rows) {
  __shared__ __attribute__((aligned(64))) __bf16 lA[2 * ASZ];   // [m][k] padded
  __shared__ __attribute__((aligned(64))) __bf16 lB1[2 * BSZ];  // [n][k] padded
  __shared__ __attribute__((aligned(64))) __bf16 lB3[2 * BSZ];

  int e = 0;
  if (tileE) { e = tileE[blockIdx.y]; if (e < 0) return; }

  const int m0 = blockIdx.y * TM;
  const int n0 = blockIdx.x * TN;
  const int tid  = threadIdx.x;
  const int lane = tid & 31;
  const int wave = tid >> 5;
  const int wm   = wave >> 1;          // 0..3 m-subtile
  const int wn16 = (wave & 1) * 16;    // 0 / 16 n base

  const unsigned ldsA = (unsigned)(unsigned long long)lA;
  const __bf16* aTile = X + (long)m0 * HH;   // advances by TK in k

  // cooperative B load: one column of 8 k-values per thread
  const int bn  = tid & 63;            // 0..63 column
  const int bkc = (tid >> 6) * 8;      // 0,8,16,24
  const float* w1p = W1base + (long)e * wstride + (long)bkc * II + n0 + bn;
  const float* w3p = W3base + (long)e * wstride + (long)bkc * II + n0 + bn;
  const int bst = bn * LDB + bkc;

  v8f c1a = {}, c1b = {}, c3a = {}, c3b = {};

  const int aoff  = (wm * 16 + (lane & 15)) * LDA + (lane >> 4) * 16;
  const int col0  = wn16 + (lane & 15);
  const int col1  = col0 + 32;
  const int kloB  = (lane >> 4) * 16;
  const int boff0 = col0 * LDB + kloB;
  const int boff1 = col1 * LDB + kloB;

  // ---- prologue: fill buffer 0 ----
  if (wave == 0) tdm_load_tile(ldsA, aTile, HH, (unsigned)rows);
  {
    float b1v[8], b3v[8];
#pragma unroll
    for (int i = 0; i < 8; ++i) { b1v[i] = w1p[(long)i * II]; b3v[i] = w3p[(long)i * II]; }
    v8bf p1, p3;
#pragma unroll
    for (int i = 0; i < 8; ++i) { p1[i] = (__bf16)b1v[i]; p3[i] = (__bf16)b3v[i]; }
    *(v8bf*)(lB1 + bst) = p1;
    *(v8bf*)(lB3 + bst) = p3;
  }
  if (wave == 0) __builtin_amdgcn_s_wait_tensorcnt(0);
  __syncthreads();

  int buf = 0;
  for (int k0 = 0; k0 < HH; k0 += TK) {
    const int nb = buf ^ 1;
    const bool more = (k0 + TK) < HH;

    // ---- issue next tile's loads (overlap with WMMAs below) ----
    float b1v[8], b3v[8];
    if (more) {
      if (wave == 0)
        tdm_load_tile(ldsA + nb * (ASZ * 2), aTile + k0 + TK, HH, (unsigned)rows);
      const float* q1 = w1p + (long)TK * II;
      const float* q3 = w3p + (long)TK * II;
#pragma unroll
      for (int i = 0; i < 8; ++i) { b1v[i] = q1[(long)i * II]; b3v[i] = q3[(long)i * II]; }
      if (k0 + 2 * TK < HH) {
        __builtin_prefetch(q1 + (long)TK * II, 0, 3);
        __builtin_prefetch(q3 + (long)TK * II, 0, 3);
      }
    }

    // ---- compute on current buffer ----
    const __bf16* A  = lA  + buf * ASZ;
    const __bf16* B1 = lB1 + buf * BSZ;
    const __bf16* B3 = lB3 + buf * BSZ;
    v8bf alo = *(const v8bf*)(A + aoff);
    v8bf ahi = *(const v8bf*)(A + aoff + 8);
    v16bf a = __builtin_shufflevector(alo, ahi, 0, 1, 2, 3, 4, 5, 6, 7,
                                      8, 9, 10, 11, 12, 13, 14, 15);
    v16bf b1x = *(const v16bf*)(B1 + boff0);
    v16bf b1y = *(const v16bf*)(B1 + boff1);
    v16bf b3x = *(const v16bf*)(B3 + boff0);
    v16bf b3y = *(const v16bf*)(B3 + boff1);
    c1a = __builtin_amdgcn_wmma_f32_16x16x32_bf16(false, a, false, b1x, (short)0, c1a, false, false);
    c1b = __builtin_amdgcn_wmma_f32_16x16x32_bf16(false, a, false, b1y, (short)0, c1b, false, false);
    c3a = __builtin_amdgcn_wmma_f32_16x16x32_bf16(false, a, false, b3x, (short)0, c3a, false, false);
    c3b = __builtin_amdgcn_wmma_f32_16x16x32_bf16(false, a, false, b3y, (short)0, c3b, false, false);

    // ---- commit next B tile to LDS ----
    if (more) {
      v8bf p1, p3;
#pragma unroll
      for (int i = 0; i < 8; ++i) { p1[i] = (__bf16)b1v[i]; p3[i] = (__bf16)b3v[i]; }
      *(v8bf*)(lB1 + nb * BSZ + bst) = p1;
      *(v8bf*)(lB3 + nb * BSZ + bst) = p3;
    }

    if (wave == 0) __builtin_amdgcn_s_wait_tensorcnt(0);
    __syncthreads();
    w1p += (long)TK * II;
    w3p += (long)TK * II;
    buf = nb;
  }

  // fused SwiGLU epilogue, store bf16 at two column groups
  const int outc0 = n0 + col0;
  const int outc1 = n0 + col1;
#pragma unroll
  for (int r = 0; r < 8; ++r) {
    int outr = m0 + wm * 16 + r + ((lane >> 4) * 8);
    float y0 = c1a[r] * fast_sigmoid(c1a[r]) * c3a[r];
    float y1 = c1b[r] * fast_sigmoid(c1b[r]) * c3b[r];
    Y[(long)outr * II + outc0] = (__bf16)y0;
    Y[(long)outr * II + outc1] = (__bf16)y1;
  }
}

// ---------------- GEMM2: Out = Y @ W2 (direct store or scatter-add) --------
__global__ __launch_bounds__(256)
void gemm2_kernel(const __bf16* __restrict__ Ybuf, const float* __restrict__ W2base,
                  float* __restrict__ Out, const int* __restrict__ tileE,
                  const int* __restrict__ perm, long wstride, int rows) {
  __shared__ __attribute__((aligned(64))) __bf16 lA[2 * ASZ];
  __shared__ __attribute__((aligned(64))) __bf16 lB[2 * BSZ];

  int e = 0;
  if (tileE) { e = tileE[blockIdx.y]; if (e < 0) return; }

  const int m0 = blockIdx.y * TM;
  const int n0 = blockIdx.x * TN;
  const int tid  = threadIdx.x;
  const int lane = tid & 31;
  const int wave = tid >> 5;
  const int wm   = wave >> 1;
  const int wn16 = (wave & 1) * 16;

  const unsigned ldsA = (unsigned)(unsigned long long)lA;
  const __bf16* aTile = Ybuf + (long)m0 * II;

  const int bn  = tid & 63;
  const int bkc = (tid >> 6) * 8;
  const float* w2p = W2base + (long)e * wstride + (long)bkc * HH + n0 + bn;
  const int bst = bn * LDB + bkc;

  v8f ca = {}, cb = {};

  const int aoff  = (wm * 16 + (lane & 15)) * LDA + (lane >> 4) * 16;
  const int col0  = wn16 + (lane & 15);
  const int col1  = col0 + 32;
  const int kloB  = (lane >> 4) * 16;
  const int boff0 = col0 * LDB + kloB;
  const int boff1 = col1 * LDB + kloB;

  // ---- prologue ----
  if (wave == 0) tdm_load_tile(ldsA, aTile, II, (unsigned)rows);
  {
    float bv[8];
#pragma unroll
    for (int i = 0; i < 8; ++i) bv[i] = w2p[(long)i * HH];
    v8bf p;
#pragma unroll
    for (int i = 0; i < 8; ++i) p[i] = (__bf16)bv[i];
    *(v8bf*)(lB + bst) = p;
  }
  if (wave == 0) __builtin_amdgcn_s_wait_tensorcnt(0);
  __syncthreads();

  int buf = 0;
  for (int k0 = 0; k0 < II; k0 += TK) {
    const int nb = buf ^ 1;
    const bool more = (k0 + TK) < II;

    float bv[8];
    if (more) {
      if (wave == 0)
        tdm_load_tile(ldsA + nb * (ASZ * 2), aTile + k0 + TK, II, (unsigned)rows);
      const float* q = w2p + (long)TK * HH;
#pragma unroll
      for (int i = 0; i < 8; ++i) bv[i] = q[(long)i * HH];
      if (k0 + 2 * TK < II)
        __builtin_prefetch(q + (long)TK * HH, 0, 3);
    }

    const __bf16* A = lA + buf * ASZ;
    const __bf16* B = lB + buf * BSZ;
    v8bf alo = *(const v8bf*)(A + aoff);
    v8bf ahi = *(const v8bf*)(A + aoff + 8);
    v16bf a = __builtin_shufflevector(alo, ahi, 0, 1, 2, 3, 4, 5, 6, 7,
                                      8, 9, 10, 11, 12, 13, 14, 15);
    v16bf bx = *(const v16bf*)(B + boff0);
    v16bf by = *(const v16bf*)(B + boff1);
    ca = __builtin_amdgcn_wmma_f32_16x16x32_bf16(false, a, false, bx, (short)0, ca, false, false);
    cb = __builtin_amdgcn_wmma_f32_16x16x32_bf16(false, a, false, by, (short)0, cb, false, false);

    if (more) {
      v8bf p;
#pragma unroll
      for (int i = 0; i < 8; ++i) p[i] = (__bf16)bv[i];
      *(v8bf*)(lB + nb * BSZ + bst) = p;
    }

    if (wave == 0) __builtin_amdgcn_s_wait_tensorcnt(0);
    __syncthreads();
    w2p += (long)TK * HH;
    buf = nb;
  }

  const int outc0 = n0 + col0;
  const int outc1 = n0 + col1;
#pragma unroll
  for (int r = 0; r < 8; ++r) {
    int row = m0 + wm * 16 + r + ((lane >> 4) * 8);
    if (perm) {
      int t = perm[row];
      if (t >= 0) {
        Out[(long)t * HH + outc0] += ca[r];   // unique (t, col) per block
        Out[(long)t * HH + outc1] += cb[r];
      }
    } else {
      Out[(long)row * HH + outc0] = ca[r];
      Out[(long)row * HH + outc1] = cb[r];
    }
  }
}

// ---------------- host-side launch ----------------
extern "C" void kernel_launch(void* const* d_in, const int* in_sizes, int n_in,
                              void* d_out, int out_size, void* d_ws, size_t ws_size,
                              hipStream_t stream) {
  const float* x   = (const float*)d_in[0];  // [TT, HH]
  const float* rw  = (const float*)d_in[1];  // [HH, EE]
  const float* sw1 = (const float*)d_in[2];  // [HH, II]
  const float* sw3 = (const float*)d_in[3];  // [HH, II]
  const float* sw2 = (const float*)d_in[4];  // [II, HH]
  const float* ew1 = (const float*)d_in[5];  // [EE, HH, II]
  const float* ew3 = (const float*)d_in[6];  // [EE, HH, II]
  const float* ew2 = (const float*)d_in[7];  // [EE, II, HH]
  float* out = (float*)d_out;                // [TT, HH]

  // workspace carve-out (256B aligned slices)
  char* base = (char*)d_ws;
  auto alloc = [&](size_t bytes) -> void* {
    void* p = (void*)base;
    base += (bytes + 255) & ~(size_t)255;
    return p;
  };
  int*    eid     = (int*)alloc((size_t)TT * 4);
  float*  wgt     = (float*)alloc((size_t)TT * 4);
  int*    counts  = (int*)alloc(64);
  int*    cursor  = (int*)alloc(64);
  int*    offsets = (int*)alloc(64);
  int*    tileE   = (int*)alloc((size_t)NMT * 4);
  int*    perm    = (int*)alloc((size_t)TPAD * 4);
  __bf16* Xsh     = (__bf16*)alloc((size_t)TT * HH * 2);
  __bf16* Xg      = (__bf16*)alloc((size_t)TPAD * HH * 2);
  __bf16* Ysh     = (__bf16*)alloc((size_t)TT * II * 2);
  __bf16* Yg      = (__bf16*)alloc((size_t)TPAD * II * 2);
  (void)ws_size; (void)in_sizes; (void)n_in; (void)out_size;

  const long wstride12 = (long)HH * II;  // expert w1/w3 stride
  const long wstride2  = (long)II * HH;  // expert w2 stride

  init_kernel<<<1, 256, 0, stream>>>(counts, cursor, tileE, perm);
  router_kernel<<<TT / 8, 256, 0, stream>>>(x, rw, eid, wgt);
  count_kernel<<<TT / 256, 256, 0, stream>>>(eid, counts);
  scan_kernel<<<1, 1, 0, stream>>>(counts, offsets, tileE);
  scatter_kernel<<<TT / 256, 256, 0, stream>>>(eid, offsets, cursor, perm);
  cvt_shared_kernel<<<TT, 256, 0, stream>>>(x, Xsh);
  gather_kernel<<<TPAD, 256, 0, stream>>>(x, perm, wgt, Xg);

  // shared expert path
  gemm1_kernel<<<dim3(II / TN, TT / TM), 256, 0, stream>>>(
      Xsh, sw1, sw3, Ysh, nullptr, 0, TT);
  // routed expert path
  gemm1_kernel<<<dim3(II / TN, NMT), 256, 0, stream>>>(
      Xg, ew1, ew3, Yg, tileE, wstride12, TPAD);

  // down-proj: shared writes, routed accumulates
  gemm2_kernel<<<dim3(HH / TN, TT / TM), 256, 0, stream>>>(
      Ysh, sw2, out, nullptr, nullptr, 0, TT);
  gemm2_kernel<<<dim3(HH / TN, NMT), 256, 0, stream>>>(
      Yg, ew2, out, tileE, perm, wstride2, TPAD);
}
